// External_Attention_67095979098144
// MI455X (gfx1250) — matrix operand order repaired and checked by reading, not compiled, not run on hardware
//
#include <hip/hip_runtime.h>

#define BB 16
#define CC 512
#define MM 256
#define NN 4096

typedef __attribute__((ext_vector_type(16))) __bf16        v16bf;
typedef __attribute__((ext_vector_type(8)))  float         v8f;
typedef __attribute__((ext_vector_type(4)))  unsigned int  v4u;

union Frag { v4u q[2]; v16bf v; };

__device__ __forceinline__ unsigned short f2bf(float f) {
  __bf16 h = (__bf16)f;                       // native v_cvt_*_bf16_f32
  return __builtin_bit_cast(unsigned short, h);
}
__device__ __forceinline__ unsigned packbf2(float a, float b) {
  return (unsigned)f2bf(a) | ((unsigned)f2bf(b) << 16);
}
__device__ __forceinline__ float bf2f(unsigned short h) {
  return __builtin_bit_cast(float, ((unsigned)h) << 16);
}

// ---------------------------------------------------------------------------
// GEMM1: S[b,m,n] (bf16 logits) = sum_c Wk[m,c] * x[b,c,n]
// Block tile 128(M) x 128(N), K-step 32, 8 waves each computing 32x64.
// ---------------------------------------------------------------------------
__global__ __launch_bounds__(256) void k_gemm1(const float* __restrict__ x,
                                               const float* __restrict__ Wk,
                                               unsigned short* __restrict__ S) {
  __shared__ unsigned short As[2][128 * 40];  // [m][k] k-contiguous, stride 40
  __shared__ unsigned short Bs[2][128 * 40];  // [n][k] k-contiguous, stride 40

  const int t  = threadIdx.x;
  const int b  = blockIdx.z;
  const int m0 = blockIdx.y * 128;
  const int n0 = blockIdx.x * 128;
  const int w = t >> 5, lane = t & 31;
  const int wm = w >> 1, wn = w & 1;          // 4 x 2 wave grid
  const int lr = lane & 15, lh = lane >> 4;

  v8f acc[2][4] = {};

  const int am  = t >> 4;              // A staging: row 0..15 (+16*i)
  const int ak  = (t & 15) << 1;       // even k
  const int bn  = t & 127;             // B staging: col
  const int bk0 = (t >> 7) << 1;       // 0 or 2

  auto stage = [&](int buf, int k0) {
#pragma unroll
    for (int i = 0; i < 8; ++i) {
      int m = am + i * 16;
      const float* wp = Wk + (size_t)(m0 + m) * CC + k0 + ak;
      *(unsigned*)&As[buf][m * 40 + ak] = packbf2(wp[0], wp[1]);
    }
#pragma unroll
    for (int i = 0; i < 8; ++i) {
      int k = bk0 + i * 4;
      const float* xp = x + ((size_t)b * CC + (k0 + k)) * NN + n0 + bn;
      *(unsigned*)&Bs[buf][bn * 40 + k] = packbf2(xp[0], xp[NN]);
      if (k0 + 32 < CC) __builtin_prefetch(xp + 32 * NN, 0, 1);  // 2 K-steps ahead
    }
  };

  stage(0, 0);
  __syncthreads();
  const int KT = CC / 32;
  for (int kt = 0; kt < KT; ++kt) {
    const int cur = kt & 1;
    if (kt + 1 < KT) stage(cur ^ 1, (kt + 1) * 32);

    Frag a[2], bf[4];
#pragma unroll
    for (int s = 0; s < 2; ++s) {
      int row = wm * 32 + s * 16 + lr;
      const unsigned short* p = &As[cur][row * 40 + lh * 8];
      a[s].q[0] = *(const v4u*)p;          // A: K = lh*8 .. +7
      a[s].q[1] = *(const v4u*)(p + 16);   // A: K = 16+lh*8 .. +7
    }
#pragma unroll
    for (int j = 0; j < 4; ++j) {
      int col = wn * 64 + j * 16 + lr;
      const unsigned short* p = &Bs[cur][col * 40 + lh * 8];
      bf[j].q[0] = *(const v4u*)p;         // B: K = lh*8 .. +7
      bf[j].q[1] = *(const v4u*)(p + 16);  // B: K = 16+lh*8 .. +7
    }
#pragma unroll
    for (int s = 0; s < 2; ++s)
#pragma unroll
      for (int j = 0; j < 4; ++j)
        acc[s][j] = __builtin_amdgcn_wmma_f32_16x16x32_bf16(
            false, a[s].v, false, bf[j].v, (short)0, acc[s][j], false, false);
    __syncthreads();
  }

#pragma unroll
  for (int s = 0; s < 2; ++s)
#pragma unroll
    for (int j = 0; j < 4; ++j) {
      int cg = n0 + wn * 64 + j * 16 + lr;
      int rw = m0 + wm * 32 + s * 16 + lh * 8;
      unsigned short* op = S + ((size_t)b * MM + rw) * NN + cg;
#pragma unroll
      for (int r = 0; r < 8; ++r) op[(size_t)r * NN] = f2bf(acc[s][j][r]);
    }
}

// ---------------------------------------------------------------------------
// Softmax over N (in place on bf16 logits). One block per (b,m) row.
// ---------------------------------------------------------------------------
__global__ __launch_bounds__(256) void k_softmax(unsigned short* __restrict__ S) {
  __shared__ float red[8];
  const int t = threadIdx.x;
  const int wid = t >> 5, lane = t & 31;
  const size_t base = (size_t)blockIdx.x * NN;

  float v[16];
  float mx = -3.4e38f;
#pragma unroll
  for (int i = 0; i < 16; ++i) {
    v[i] = bf2f(S[base + t + i * 256]);
    mx = fmaxf(mx, v[i]);
  }
#pragma unroll
  for (int off = 16; off; off >>= 1) mx = fmaxf(mx, __shfl_xor(mx, off, 32));
  if (lane == 0) red[wid] = mx;
  __syncthreads();
#pragma unroll
  for (int i = 0; i < 8; ++i) mx = fmaxf(mx, red[i]);

  float s = 0.f;
#pragma unroll
  for (int i = 0; i < 16; ++i) {
    v[i] = __expf(v[i] - mx);
    s += v[i];
  }
#pragma unroll
  for (int off = 16; off; off >>= 1) s += __shfl_xor(s, off, 32);
  __syncthreads();                 // red reuse
  if (lane == 0) red[wid] = s;
  __syncthreads();
  float tot = 0.f;
#pragma unroll
  for (int i = 0; i < 8; ++i) tot += red[i];
  const float r = 1.0f / tot;
#pragma unroll
  for (int i = 0; i < 16; ++i) S[base + t + i * 256] = f2bf(v[i] * r);
}

// ---------------------------------------------------------------------------
// inv[b,n] = 1 / (1e-9 + sum_m softattn[b,m,n])
// ---------------------------------------------------------------------------
__global__ __launch_bounds__(256) void k_colsum(const unsigned short* __restrict__ S,
                                                float* __restrict__ inv) {
  const int gid = blockIdx.x * 256 + threadIdx.x;   // 0 .. B*N-1
  const int b = gid >> 12;                          // N = 4096
  const int n = gid & (NN - 1);
  const unsigned short* p = S + ((size_t)b * MM) * NN + n;
  float s = 0.f;
  for (int m = 0; m < MM; ++m) s += bf2f(p[(size_t)m * NN]);
  inv[gid] = 1.0f / (1e-9f + s);
}

// ---------------------------------------------------------------------------
// GEMM2: out[b,c,n] = x[b,c,n] + inv[b,n] * sum_m Wv[c,m] * S[b,m,n]
// ---------------------------------------------------------------------------
__global__ __launch_bounds__(256) void k_gemm2(const float* __restrict__ x,
                                               const float* __restrict__ Wv,
                                               const unsigned short* __restrict__ S,
                                               const float* __restrict__ inv,
                                               float* __restrict__ out) {
  __shared__ unsigned short As[2][128 * 40];  // Wv tile [c][k(m)]
  __shared__ unsigned short Bs[2][128 * 40];  // S tile  [n][k(m)]

  const int t  = threadIdx.x;
  const int b  = blockIdx.z;
  const int c0 = blockIdx.y * 128;
  const int n0 = blockIdx.x * 128;
  const int w = t >> 5, lane = t & 31;
  const int wm = w >> 1, wn = w & 1;
  const int lr = lane & 15, lh = lane >> 4;

  v8f acc[2][4] = {};

  const int am = t >> 4;
  const int ak = (t & 15) << 1;
  const int n2 = (t & 63) << 1;
  const int kr = t >> 6;          // 0..3

  auto stage = [&](int buf, int k0) {
#pragma unroll
    for (int i = 0; i < 8; ++i) {
      int c = am + i * 16;
      const float* wp = Wv + (size_t)(c0 + c) * MM + k0 + ak;
      *(unsigned*)&As[buf][c * 40 + ak] = packbf2(wp[0], wp[1]);
    }
#pragma unroll
    for (int i = 0; i < 8; ++i) {
      int k = kr + i * 4;
      const unsigned short* sp = S + ((size_t)b * MM + k0 + k) * NN + n0 + n2;
      unsigned u = *(const unsigned*)sp;
      Bs[buf][n2 * 40 + k]       = (unsigned short)u;
      Bs[buf][(n2 + 1) * 40 + k] = (unsigned short)(u >> 16);
      if (k0 + 32 < MM) __builtin_prefetch(sp + 32 * NN, 0, 1);  // 2 K-steps ahead
    }
  };

  stage(0, 0);
  __syncthreads();
  const int KT = MM / 32;
  for (int kt = 0; kt < KT; ++kt) {
    const int cur = kt & 1;
    if (kt + 1 < KT) stage(cur ^ 1, (kt + 1) * 32);

    Frag a[2], bf[4];
#pragma unroll
    for (int s = 0; s < 2; ++s) {
      int row = wm * 32 + s * 16 + lr;
      const unsigned short* p = &As[cur][row * 40 + lh * 8];
      a[s].q[0] = *(const v4u*)p;
      a[s].q[1] = *(const v4u*)(p + 16);
    }
#pragma unroll
    for (int j = 0; j < 4; ++j) {
      int col = wn * 64 + j * 16 + lr;
      const unsigned short* p = &Bs[cur][col * 40 + lh * 8];
      bf[j].q[0] = *(const v4u*)p;
      bf[j].q[1] = *(const v4u*)(p + 16);
    }
#pragma unroll
    for (int s = 0; s < 2; ++s)
#pragma unroll
      for (int j = 0; j < 4; ++j)
        acc[s][j] = __builtin_amdgcn_wmma_f32_16x16x32_bf16(
            false, a[s].v, false, bf[j].v, (short)0, acc[s][j], false, false);
    __syncthreads();
  }

#pragma unroll
  for (int s = 0; s < 2; ++s)
#pragma unroll
    for (int j = 0; j < 4; ++j) {
      int cg = n0 + wn * 64 + j * 16 + lr;
      float iv = inv[b * NN + cg];
      int rc = c0 + wm * 32 + s * 16 + lh * 8;
      size_t idx = ((size_t)b * CC + rc) * NN + cg;
#pragma unroll
      for (int r = 0; r < 8; ++r)
        out[idx + (size_t)r * NN] = x[idx + (size_t)r * NN] + acc[s][j][r] * iv;
    }
}

// ---------------------------------------------------------------------------
extern "C" void kernel_launch(void* const* d_in, const int* in_sizes, int n_in,
                              void* d_out, int out_size, void* d_ws, size_t ws_size,
                              hipStream_t stream) {
  const float* x  = (const float*)d_in[0];
  const float* Wk = (const float*)d_in[1];
  const float* Wv = (const float*)d_in[2];
  float* out = (float*)d_out;

  unsigned short* S = (unsigned short*)d_ws;                       // bf16 [B,M,N]
  float* inv = (float*)((char*)d_ws +
                        (size_t)BB * MM * NN * sizeof(unsigned short));  // [B,N]

  k_gemm1  <<<dim3(NN / 128, MM / 128, BB), 256, 0, stream>>>(x, Wk, S);
  k_softmax<<<dim3(BB * MM),                256, 0, stream>>>(S);
  k_colsum <<<dim3((BB * NN) / 256),        256, 0, stream>>>(S, inv);
  k_gemm2  <<<dim3(NN / 128, CC / 128, BB), 256, 0, stream>>>(x, Wv, S, inv, out);
}
